// IrabHead_66898410602527
// MI455X (gfx1250) — compile-verified, matrix operand order checked by reading, not run on hardware
//
#include <hip/hip_runtime.h>
#include <hip/hip_bf16.h>

typedef _Float16 v16h __attribute__((ext_vector_type(16)));
typedef _Float16 v8h  __attribute__((ext_vector_type(8)));
typedef _Float16 v4h  __attribute__((ext_vector_type(4)));
typedef float    v8f  __attribute__((ext_vector_type(8)));

#define B_   32
#define L_   2048
#define H_   768
#define W_   512
#define NC_  16
#define HD2  384              // H/2
#define M_   (B_ * W_)        // 16384 rows
#define MT   32               // M-tile per block (2 WMMA M-subtiles)

// ---------------------------------------------------------------------------
// Kernel 1: ragged span mean-pool -> f16 pooled matrix (M_ x H_), plus mask.
// One block per span; 192 threads, each owns one float4 column group (192*4=768).
// ---------------------------------------------------------------------------
__global__ __launch_bounds__(192)
void pool_kernel(const float* __restrict__ enc,
                 const int* __restrict__ starts,
                 const int* __restrict__ ends,
                 _Float16* __restrict__ pooled,
                 float* __restrict__ mask) {
    const int s  = blockIdx.x;          // span id in [0, M_)
    const int b  = s >> 9;              // / 512
    const int w  = s & (W_ - 1);
    const int st = starts[b * W_ + w];
    const int en = ends[b * W_ + w];
    const bool valid = (st < L_) && (en <= L_) && (st < en);

    const int hoff = threadIdx.x * 4;
    float4 acc = make_float4(0.f, 0.f, 0.f, 0.f);
    if (valid) {
        const float* base = enc + ((size_t)b * L_ + (size_t)st) * H_ + hoff;
        const int cnt = en - st;        // in [1, 8]
        #pragma unroll 1
        for (int r = 0; r < cnt; ++r) {
            float4 v = *(const float4*)(base + (size_t)r * H_);
            acc.x += v.x; acc.y += v.y; acc.z += v.z; acc.w += v.w;
        }
        const float inv = 1.0f / (float)cnt;
        acc.x *= inv; acc.y *= inv; acc.z *= inv; acc.w *= inv;
    }
    v4h h4;
    h4[0] = (_Float16)acc.x; h4[1] = (_Float16)acc.y;
    h4[2] = (_Float16)acc.z; h4[3] = (_Float16)acc.w;
    *(v4h*)(pooled + (size_t)s * H_ + hoff) = h4;

    if (threadIdx.x == 0) mask[s] = valid ? 1.0f : 0.0f;
}

// ---------------------------------------------------------------------------
// Kernel 2: W1 (768 x 384, k-major f32) -> W1t (384 x 768, n-major f16)
// so WMMA B-fragments are two contiguous 16B loads per lane.
// ---------------------------------------------------------------------------
__global__ __launch_bounds__(256)
void transpose_w1(const float* __restrict__ W1, _Float16* __restrict__ Bt) {
    const int idx = blockIdx.x * 256 + threadIdx.x;
    if (idx >= H_ * HD2) return;
    const int k = idx / HD2;
    const int n = idx - k * HD2;
    Bt[(size_t)n * H_ + k] = (_Float16)W1[idx];
}

// ---------------------------------------------------------------------------
// Kernel 3: fused GEMM1 (f16 WMMA, f32 accum) + bias + exact GELU + GEMM2.
// Block = 32 rows x 384 cols of h. 8 waves; wave w owns N-subtiles 3w..3w+2
// and both 16-row M-subtiles (6 accumulators). Each B fragment feeds 2 WMMAs,
// each A fragment feeds 3 -> 10 b128 loads per 6 v_wmma_f32_16x16x32_f16.
// ---------------------------------------------------------------------------
__device__ __forceinline__ v16h load_frag16(const _Float16* row, int off_lo, int off_hi) {
    v8h lo = *(const v8h*)(row + off_lo);
    v8h hi = *(const v8h*)(row + off_hi);
    return __builtin_shufflevector(lo, hi, 0,1,2,3,4,5,6,7,8,9,10,11,12,13,14,15);
}

__global__ __launch_bounds__(256)
void gemm_fused(const _Float16* __restrict__ Af,   // M_ x H_ (row-major)
                const _Float16* __restrict__ Bt,   // HD2 x H_ (n-major)
                const float* __restrict__ bias1,   // HD2
                const float* __restrict__ W2,      // HD2 x NC_ (k-major)
                const float* __restrict__ bias2,   // NC_
                float* __restrict__ logits) {      // M_ x NC_
    constexpr int S = HD2 + 4;                     // LDS row stride (pad vs bank conflicts)
    __shared__ float hbuf[MT * S];                 // 32 x 388 f32 ~= 50 KB

    const int m0   = blockIdx.x * MT;
    const int lane = threadIdx.x & 31;
    const int wave = threadIdx.x >> 5;
    const int mrow = lane & 15;                    // A/B: M (or N) index per lane
    const int half = lane >> 4;                    // lane half selects K sub-range

    const _Float16* arow[2];
    arow[0] = Af + (size_t)(m0 + mrow) * H_;
    arow[1] = Af + (size_t)(m0 + 16 + mrow) * H_;
    const _Float16* brow[3];
    #pragma unroll
    for (int t = 0; t < 3; ++t)
        brow[t] = Bt + (size_t)(wave * 48 + t * 16 + mrow) * H_;

    v8f acc[2][3] = {};

    #pragma unroll 2
    for (int k0 = 0; k0 < H_; k0 += 32) {
        // ISA 16-bit fragment layout: VGPR0-3 <- K=k0+half*8..+7, VGPR4-7 <- K=k0+16+half*8..+7
        const int off_lo = k0 + half * 8;
        const int off_hi = k0 + 16 + half * 8;
        v16h a0 = load_frag16(arow[0], off_lo, off_hi);
        v16h a1 = load_frag16(arow[1], off_lo, off_hi);
        #pragma unroll
        for (int t = 0; t < 3; ++t) {
            v16h bfrag = load_frag16(brow[t], off_lo, off_hi);
            acc[0][t] = __builtin_amdgcn_wmma_f32_16x16x32_f16(
                false, a0, false, bfrag, (short)0, acc[0][t], false, false);
            acc[1][t] = __builtin_amdgcn_wmma_f32_16x16x32_f16(
                false, a1, false, bfrag, (short)0, acc[1][t], false, false);
        }
    }

    // Epilogue: bias + exact GELU, spill to LDS.
    // C/D layout: VGPR r -> M = r + 8*half (+16*msub), N = lane&15 (+16*nsub).
    #pragma unroll
    for (int mi = 0; mi < 2; ++mi) {
        #pragma unroll
        for (int t = 0; t < 3; ++t) {
            const int n = wave * 48 + t * 16 + mrow;
            const float bv = bias1[n];
            #pragma unroll
            for (int r = 0; r < 8; ++r) {
                const int m = mi * 16 + r + 8 * half;
                float x = acc[mi][t][r] + bv;
                float g = 0.5f * x * (1.0f + erff(x * 0.70710678118654752f));
                hbuf[m * S + n] = g;
            }
        }
    }
    __syncthreads();

    // GEMM2: each thread handles 2 (row, class) outputs. W2 reads coalesce
    // across the 16 class-threads of a row; hbuf value broadcasts in-group.
    const int mA = threadIdx.x >> 4;               // rows 0..15
    const int mB = mA + 16;                        // rows 16..31
    const int c  = threadIdx.x & 15;
    float sumA = bias2[c];
    float sumB = sumA;
    #pragma unroll 8
    for (int j = 0; j < HD2; ++j) {
        const float wv = W2[j * NC_ + c];
        sumA = fmaf(hbuf[mA * S + j], wv, sumA);
        sumB = fmaf(hbuf[mB * S + j], wv, sumB);
    }
    logits[((size_t)(m0 + mA)) * NC_ + c] = sumA;
    logits[((size_t)(m0 + mB)) * NC_ + c] = sumB;
}

// ---------------------------------------------------------------------------
extern "C" void kernel_launch(void* const* d_in, const int* in_sizes, int n_in,
                              void* d_out, int out_size, void* d_ws, size_t ws_size,
                              hipStream_t stream) {
    const float* enc    = (const float*)d_in[0];   // (B, L, H) f32
    const int*   starts = (const int*)d_in[1];     // (B, W) i32
    const int*   ends   = (const int*)d_in[2];     // (B, W) i32
    const float* W1     = (const float*)d_in[3];   // (H, H/2) f32
    const float* b1     = (const float*)d_in[4];   // (H/2,) f32
    const float* W2     = (const float*)d_in[5];   // (H/2, NC) f32
    const float* b2     = (const float*)d_in[6];   // (NC,) f32

    float* logits = (float*)d_out;                       // M_ * NC_
    float* mask   = (float*)d_out + (size_t)M_ * NC_;    // M_

    // Workspace: pooled f16 (M_ x H_) then W1^T f16 (HD2 x H_); ~25.8 MB total.
    _Float16* pooled = (_Float16*)d_ws;
    _Float16* Bt     = pooled + (size_t)M_ * H_;

    pool_kernel<<<M_, 192, 0, stream>>>(enc, starts, ends, pooled, mask);
    transpose_w1<<<(H_ * HD2 + 255) / 256, 256, 0, stream>>>(W1, Bt);
    gemm_fused<<<M_ / MT, 256, 0, stream>>>(pooled, Bt, b1, W2, b2, logits);
}
